// RSE_22428319220430
// MI455X (gfx1250) — matrix operand order repaired and checked by reading, not compiled, not run on hardware
//
#include <hip/hip_runtime.h>
#include <hip/hip_bf16.h>
#include <math.h>

// ---- model constants (match reference) ----
#define D_MODEL   768
#define D_STATE   128
#define D_CONV    4
#define HEADDIM   64
#define NHEADS    24
#define D_INNER   1536
#define CONV_DIM  1792
#define D_IN_PROJ 3352
#define BATCH     4
#define SEQ       2048
#define NTOK      (BATCH * SEQ)   // 8192
#define EPS       1e-5f

// ---- chunked-scan parameters ----
#define Q       128               // chunk length
#define NCHUNK  (SEQ / Q)         // 16

typedef __attribute__((ext_vector_type(16))) __bf16 v16bf;
typedef __attribute__((ext_vector_type(8)))  float  v8f;

// ---- fp32 <-> bf16 bit helpers (RNE) ----
__device__ __forceinline__ unsigned short f2bf(float f) {
    unsigned int u = __float_as_uint(f);
    u += 0x7FFFu + ((u >> 16) & 1u);
    return (unsigned short)(u >> 16);
}
__device__ __forceinline__ float bf2f(unsigned short b) {
    return __uint_as_float(((unsigned int)b) << 16);
}
__device__ __forceinline__ void split_store(unsigned short* hi, unsigned short* lo, float f) {
    const unsigned short h = f2bf(f);
    *hi = h;
    *lo = f2bf(f - bf2f(h));
}

// ---- WMMA fragment helpers (wave32 layouts per CDNA5 ISA) ----
struct BF16Frag { union { unsigned short u[16]; v16bf v; }; };

// A operand: element j <-> (M = row, K = (j<8?0:16) + hl*8 + (j&7)); array is [M][K]
__device__ __forceinline__ void buildA(BF16Frag& f, const unsigned short (*M)[34], int row, int hl) {
#pragma unroll
    for (int j = 0; j < 16; j++)
        f.u[j] = M[row][((j < 8) ? 0 : 16) + hl * 8 + (j & 7)];
}
// B operand: element j <-> (N = row, K = (j<8?0:16) + hl*8 + (j&7)); array is [N][K]
__device__ __forceinline__ void buildB(BF16Frag& f, const unsigned short (*M)[34], int row, int hl) {
#pragma unroll
    for (int j = 0; j < 16; j++)
        f.u[j] = M[row][((j < 8) ? 0 : 16) + hl * 8 + (j & 7)];
}
// acc += Ahi*Bhi + Ahi*Blo + Alo*Bhi   (bf16x3 ~ fp32 accuracy)
__device__ __forceinline__ void wmma_bf16x3(v8f& acc, const BF16Frag& ah, const BF16Frag& al,
                                            const BF16Frag& bh, const BF16Frag& bl) {
    acc = __builtin_amdgcn_wmma_f32_16x16x32_bf16(false, ah.v, false, bh.v, (short)0, acc, false, false);
    acc = __builtin_amdgcn_wmma_f32_16x16x32_bf16(false, ah.v, false, bl.v, (short)0, acc, false, false);
    acc = __builtin_amdgcn_wmma_f32_16x16x32_bf16(false, al.v, false, bh.v, (short)0, acc, false, false);
}

union F8 { float f[8]; v8f v; };

// =====================================================================
// Generic GEMM: Out[M,N] = scale * A[M,K] @ W[N,K]^T  (row-major fp32)
// 32x64 block tile, 8 waves of 16x16. K%32==0, M%32==0, N ragged.
// =====================================================================
__global__ void __launch_bounds__(256)
gemm_bf16x3_wmma(const float* __restrict__ A,
                 const float* __restrict__ W,
                 float* __restrict__ Out,
                 int N, int K,
                 const float* __restrict__ gate)   // non-null: scale = sigmoid(*gate)
{
    __shared__ unsigned short Ahi[32][34], Alo[32][34];
    __shared__ unsigned short Whi[64][34], Wlo[64][34];

    const int tid  = threadIdx.x;
    const int lane = tid & 31;
    const int wave = tid >> 5;
    const int wm   = wave & 1;
    const int wn   = wave >> 1;
    const int hl   = lane >> 4;
    const int l16  = lane & 15;

    const int m0 = blockIdx.y * 32;
    const int n0 = blockIdx.x * 64;

    F8 acc;
#pragma unroll
    for (int i = 0; i < 8; i++) acc.f[i] = 0.f;

    for (int k0 = 0; k0 < K; k0 += 32) {
        {   // stage A slab: 32x32, 4 elems/thread
            const int e = tid * 4;
            const int r = e >> 5, cc = e & 31;
            const float4 v = *(const float4*)(A + (size_t)(m0 + r) * K + k0 + cc);
            const float fv[4] = {v.x, v.y, v.z, v.w};
#pragma unroll
            for (int j = 0; j < 4; j++) split_store(&Ahi[r][cc + j], &Alo[r][cc + j], fv[j]);
        }
        {   // stage W slab: 64x32, 8 elems/thread
            const int e = tid * 8;
            const int r = e >> 5, cc = e & 31;
            const int gn = n0 + r;
            float fv[8];
            if (gn < N) {
                const float4 v0 = *(const float4*)(W + (size_t)gn * K + k0 + cc);
                const float4 v1 = *(const float4*)(W + (size_t)gn * K + k0 + cc + 4);
                fv[0]=v0.x; fv[1]=v0.y; fv[2]=v0.z; fv[3]=v0.w;
                fv[4]=v1.x; fv[5]=v1.y; fv[6]=v1.z; fv[7]=v1.w;
                if (k0 + 32 < K)
                    __builtin_prefetch(W + (size_t)gn * K + k0 + 32 + cc, 0, 1);
            } else {
#pragma unroll
                for (int j = 0; j < 8; j++) fv[j] = 0.f;
            }
#pragma unroll
            for (int j = 0; j < 8; j++) split_store(&Whi[r][cc + j], &Wlo[r][cc + j], fv[j]);
        }
        __syncthreads();

        BF16Frag ah, al, bh, bl;
        buildA(ah, Ahi, wm * 16 + l16, hl);
        buildA(al, Alo, wm * 16 + l16, hl);
        buildB(bh, Whi, wn * 16 + l16, hl);
        buildB(bl, Wlo, wn * 16 + l16, hl);
        wmma_bf16x3(acc.v, ah, al, bh, bl);
        __syncthreads();
    }

    float scale = 1.0f;
    if (gate) scale = 1.0f / (1.0f + __expf(-gate[0]));
    const int col = n0 + wn * 16 + l16;
    if (col < N) {
#pragma unroll
        for (int r = 0; r < 8; r++) {
            const int row = m0 + wm * 16 + hl * 8 + r;
            Out[(size_t)row * N + col] = acc.f[r] * scale;
        }
    }
}

// =====================================================================
// dt = softplus(dt_raw + dt_bias);  ldA = dt * (-exp(A_log))
// =====================================================================
__global__ void dt_da_kernel(const float* __restrict__ zx,
                             const float* __restrict__ dt_bias,
                             const float* __restrict__ A_log,
                             float* __restrict__ dtv, float* __restrict__ ldA)
{
    const int idx = blockIdx.x * blockDim.x + threadIdx.x;
    if (idx >= NTOK * NHEADS) return;
    const int h   = idx % NHEADS;
    const int tok = idx / NHEADS;
    const float x  = zx[(size_t)tok * D_IN_PROJ + D_INNER + CONV_DIM + h] + dt_bias[h];
    const float sp = (x > 20.f) ? x : log1pf(__expf(x));
    dtv[idx] = sp;
    ldA[idx] = sp * (-__expf(A_log[h]));   // log(dA), <= 0
}

// =====================================================================
// causal depthwise conv1d (k=4) + bias + silu on the xBC slice
// =====================================================================
__global__ void conv_silu_kernel(const float* __restrict__ zx,
                                 const float* __restrict__ conv_w,
                                 const float* __restrict__ conv_b,
                                 float* __restrict__ xBC)
{
    const int idx = blockIdx.x * blockDim.x + threadIdx.x;
    if (idx >= NTOK * CONV_DIM) return;
    const int cch = idx % CONV_DIM;
    const int tok = idx / CONV_DIM;
    const int l   = tok % SEQ;
    float acc = conv_b[cch];
#pragma unroll
    for (int k = 0; k < D_CONV; k++) {
        const int t = l - (D_CONV - 1) + k;
        if (t >= 0)
            acc += zx[(size_t)(tok - (D_CONV - 1) + k) * D_IN_PROJ + D_INNER + cch]
                   * conv_w[cch * D_CONV + k];
    }
    xBC[idx] = acc / (1.f + __expf(-acc));   // silu
}

// =====================================================================
// inclusive cumsum of ldA within each chunk: cum[(bh*NCHUNK+c)*Q + t]
// =====================================================================
__global__ void chunk_cumsum_kernel(const float* __restrict__ ldA, float* __restrict__ cum)
{
    const int idx = blockIdx.x * blockDim.x + threadIdx.x;  // (bh*NCHUNK + c)
    if (idx >= BATCH * NHEADS * NCHUNK) return;
    const int c  = idx % NCHUNK;
    const int bh = idx / NCHUNK;
    const int h  = bh % NHEADS;
    const int b  = bh / NHEADS;
    const size_t tok0 = (size_t)b * SEQ + (size_t)c * Q;
    float s = 0.f;
    for (int t = 0; t < Q; t++) {
        s += ldA[(tok0 + t) * NHEADS + h];
        cum[(size_t)idx * Q + t] = s;
    }
}

// =====================================================================
// per-chunk state contribution (WMMA):
//   Spart[blk][p][n] = sum_s  x[s,p] * dt_s * exp(T - cum_s) * B[s,n]
// blk = bh*NCHUNK + c ; tile 64x128, K = Q
// =====================================================================
__global__ void __launch_bounds__(256)
chunk_state_kernel(const float* __restrict__ xBC, const float* __restrict__ dtv,
                   const float* __restrict__ cum, float* __restrict__ Sbuf)
{
    __shared__ unsigned short XThi[64][34],  XTlo[64][34];    // A: [p][s]
    __shared__ unsigned short BThi[128][34], BTlo[128][34];   // B: [n][s]
    __shared__ float cumc[Q];

    const int blk = blockIdx.x;
    const int c   = blk % NCHUNK;
    const int bh  = blk / NCHUNK;
    const int h   = bh % NHEADS;
    const int b   = bh / NHEADS;
    const size_t tok0 = (size_t)b * SEQ + (size_t)c * Q;

    const int tid  = threadIdx.x;
    const int lane = tid & 31, wave = tid >> 5;
    const int hl = lane >> 4, l16 = lane & 15;
    const int mt  = wave & 3;            // p-tile (0..3)
    const int ng4 = (wave >> 2) * 4;     // first of 4 n-tiles

    if (tid < Q) cumc[tid] = cum[(size_t)blk * Q + tid];
    __syncthreads();
    const float T = cumc[Q - 1];

    F8 acc[4];
#pragma unroll
    for (int i = 0; i < 4; i++)
#pragma unroll
        for (int r = 0; r < 8; r++) acc[i].f[r] = 0.f;

    for (int s0 = 0; s0 < Q; s0 += 32) {
        {   // stage XT[p][s] = x[s][p] * dt_s * exp(T - cum_s)   (8 elems/thread)
            const int sl = tid >> 3;
            const int p0 = (tid & 7) * 8;
            const size_t row = tok0 + s0 + sl;
            const float coef = dtv[row * NHEADS + h] * __expf(T - cumc[s0 + sl]);
            const float4 v0 = *(const float4*)(xBC + row * CONV_DIM + h * HEADDIM + p0);
            const float4 v1 = *(const float4*)(xBC + row * CONV_DIM + h * HEADDIM + p0 + 4);
            const float fv[8] = {v0.x, v0.y, v0.z, v0.w, v1.x, v1.y, v1.z, v1.w};
#pragma unroll
            for (int j = 0; j < 8; j++)
                split_store(&XThi[p0 + j][sl], &XTlo[p0 + j][sl], fv[j] * coef);
        }
        {   // stage BT[n][s] = B[s][n]   (16 elems/thread)
            const int sl = tid >> 3;
            const int n0 = (tid & 7) * 16;
            const float* src = xBC + (tok0 + s0 + sl) * CONV_DIM + D_INNER + n0;
#pragma unroll
            for (int q4 = 0; q4 < 4; q4++) {
                const float4 v = *(const float4*)(src + q4 * 4);
                const float fv[4] = {v.x, v.y, v.z, v.w};
#pragma unroll
                for (int j = 0; j < 4; j++)
                    split_store(&BThi[n0 + q4 * 4 + j][sl], &BTlo[n0 + q4 * 4 + j][sl], fv[j]);
            }
        }
        __syncthreads();

        BF16Frag ah, al, bh2, bl2;
        buildA(ah, XThi, mt * 16 + l16, hl);
        buildA(al, XTlo, mt * 16 + l16, hl);
#pragma unroll
        for (int i = 0; i < 4; i++) {
            buildB(bh2, BThi, (ng4 + i) * 16 + l16, hl);
            buildB(bl2, BTlo, (ng4 + i) * 16 + l16, hl);
            wmma_bf16x3(acc[i].v, ah, al, bh2, bl2);
        }
        __syncthreads();
    }
#pragma unroll
    for (int i = 0; i < 4; i++) {
        const int n = (ng4 + i) * 16 + l16;
#pragma unroll
        for (int r = 0; r < 8; r++) {
            const int p = mt * 16 + hl * 8 + r;
            Sbuf[((size_t)blk * HEADDIM + p) * D_STATE + n] = acc[i].f[r];
        }
    }
}

// =====================================================================
// serial inter-chunk recurrence (only NCHUNK=16 steps), in-place:
//   tmp = Sbuf[c];  Sbuf[c] = S_run (= S_prev);  S_run = S_run*e^{T_c} + tmp
// =====================================================================
__global__ void __launch_bounds__(256)
state_recurrence_kernel(const float* __restrict__ cum, float* __restrict__ Sbuf)
{
    const int bh  = blockIdx.x;      // 0..95
    const int tid = threadIdx.x;
    float S[32];
#pragma unroll
    for (int i = 0; i < 32; i++) S[i] = 0.f;
    for (int c = 0; c < NCHUNK; c++) {
        const float eT = __expf(cum[((size_t)bh * NCHUNK + c) * Q + (Q - 1)]);
        float* base = Sbuf + ((size_t)bh * NCHUNK + c) * (HEADDIM * D_STATE);
#pragma unroll
        for (int i = 0; i < 32; i++) {
            const int e = i * 256 + tid;
            const float part = base[e];
            base[e] = S[i];
            S[i] = S[i] * eT + part;
        }
    }
}

// ---- C-slab staging helper for chunk_output ----
__device__ __forceinline__ void stage_C(unsigned short (*hi)[34], unsigned short (*lo)[34],
                                        const float* __restrict__ xBC, size_t tok0, int kn, int tid)
{
    const int t  = tid >> 1;
    const int c0 = (tid & 1) * 16;
    const float* src = xBC + (tok0 + t) * CONV_DIM + (D_INNER + D_STATE) + kn + c0;
#pragma unroll
    for (int q4 = 0; q4 < 4; q4++) {
        const float4 v = *(const float4*)(src + q4 * 4);
        const float fv[4] = {v.x, v.y, v.z, v.w};
#pragma unroll
        for (int j = 0; j < 4; j++)
            split_store(&hi[t][c0 + q4 * 4 + j], &lo[t][c0 + q4 * 4 + j], fv[j]);
    }
}

// =====================================================================
// per-chunk output (WMMA):
//   Y = e^{cum_t} * (C @ Sprev^T)                         (inter-chunk)
//     + mask_{t>=s}(C@B^T * e^{cum_t-cum_s} * dt_s) @ X   (intra-chunk)
//     + D * x
// wave w owns t-strip [16w, 16w+16); Q=128 rows, 64 cols (p)
// =====================================================================
__global__ void __launch_bounds__(256)
chunk_output_kernel(const float* __restrict__ xBC, const float* __restrict__ dtv,
                    const float* __restrict__ cum, const float* __restrict__ Sbuf,
                    const float* __restrict__ Dp, float* __restrict__ y)
{
    __shared__ unsigned short Chi[128][34],  Clo[128][34];    // A: C[t][n]
    __shared__ unsigned short Ghi[128][34],  Glo[128][34];    // A: G[t][s]
    __shared__ unsigned short SBhi[64][34],  SBlo[64][34];    // B: Sprev[p][n]
    __shared__ unsigned short XThi[64][34],  XTlo[64][34];    // B: X^T[p][s]
    __shared__ unsigned short Bshi[32][34],  Bslo[32][34];    // B: B[s][n]
    __shared__ float cumc[Q], dtc[Q];

    const int blk = blockIdx.x;
    const int c   = blk % NCHUNK;
    const int bh  = blk / NCHUNK;
    const int h   = bh % NHEADS;
    const int b   = bh / NHEADS;
    const size_t tok0 = (size_t)b * SEQ + (size_t)c * Q;

    const int tid  = threadIdx.x;
    const int lane = tid & 31, wave = tid >> 5;
    const int hl = lane >> 4, l16 = lane & 15;

    if (tid < Q) {
        cumc[tid] = cum[(size_t)blk * Q + tid];
        dtc[tid]  = dtv[(tok0 + tid) * NHEADS + h];
    }
    __syncthreads();

    F8 accY[4];
#pragma unroll
    for (int i = 0; i < 4; i++)
#pragma unroll
        for (int r = 0; r < 8; r++) accY[i].f[r] = 0.f;

    // ---------- phase 1: Y_inter = C @ Sprev^T  (K over n) ----------
    for (int kn = 0; kn < D_STATE; kn += 32) {
        stage_C(Chi, Clo, xBC, tok0, kn, tid);
        {   // stage Sprev slab [p][kn..kn+31]  (8 elems/thread)
            const int p  = tid >> 2;
            const int c0 = (tid & 3) * 8;
            const float* src = Sbuf + ((size_t)blk * HEADDIM + p) * D_STATE + kn + c0;
            const float4 v0 = *(const float4*)(src);
            const float4 v1 = *(const float4*)(src + 4);
            const float fv[8] = {v0.x, v0.y, v0.z, v0.w, v1.x, v1.y, v1.z, v1.w};
#pragma unroll
            for (int j = 0; j < 8; j++)
                split_store(&SBhi[p][c0 + j], &SBlo[p][c0 + j], fv[j]);
        }
        __syncthreads();
        BF16Frag ah, al, bh2, bl2;
        buildA(ah, Chi, wave * 16 + l16, hl);
        buildA(al, Clo, wave * 16 + l16, hl);
#pragma unroll
        for (int nt = 0; nt < 4; nt++) {
            buildB(bh2, SBhi, nt * 16 + l16, hl);
            buildB(bl2, SBlo, nt * 16 + l16, hl);
            wmma_bf16x3(accY[nt].v, ah, al, bh2, bl2);
        }
        __syncthreads();
    }
    // scale inter-chunk term by e^{cum_t}
#pragma unroll
    for (int nt = 0; nt < 4; nt++)
#pragma unroll
        for (int r = 0; r < 8; r++) {
            const int t = wave * 16 + hl * 8 + r;
            accY[nt].f[r] *= __expf(cumc[t]);
        }

    // ---------- phase 2: masked intra-chunk, s-slabs of 32 ----------
    for (int ss = 0; ss < Q; ss += 32) {
        F8 accG[2];
#pragma unroll
        for (int i = 0; i < 2; i++)
#pragma unroll
            for (int r = 0; r < 8; r++) accG[i].f[r] = 0.f;

        // G_slab[t][s] = C[t,:] . B[s,:]   (K over n)
        for (int kn = 0; kn < D_STATE; kn += 32) {
            stage_C(Chi, Clo, xBC, tok0, kn, tid);
            {   // stage B rows ss..ss+31, cols kn..kn+31  (4 elems/thread)
                const int sl = tid >> 3;
                const int n0 = (tid & 7) * 4;
                const float4 v = *(const float4*)(xBC + (tok0 + ss + sl) * CONV_DIM
                                                  + D_INNER + kn + n0);
                const float fv[4] = {v.x, v.y, v.z, v.w};
#pragma unroll
                for (int j = 0; j < 4; j++)
                    split_store(&Bshi[sl][n0 + j], &Bslo[sl][n0 + j], fv[j]);
            }
            __syncthreads();
            BF16Frag ah, al, bh2, bl2;
            buildA(ah, Chi, wave * 16 + l16, hl);
            buildA(al, Clo, wave * 16 + l16, hl);
#pragma unroll
            for (int st = 0; st < 2; st++) {
                buildB(bh2, Bshi, st * 16 + l16, hl);
                buildB(bl2, Bslo, st * 16 + l16, hl);
                wmma_bf16x3(accG[st].v, ah, al, bh2, bl2);
            }
            __syncthreads();
        }
        // causal mask + decay + dt; each wave writes its own t-strip of G
#pragma unroll
        for (int st = 0; st < 2; st++)
#pragma unroll
            for (int r = 0; r < 8; r++) {
                const int t  = wave * 16 + hl * 8 + r;
                const int sg = ss + st * 16 + l16;
                float gv = 0.f;
                if (t >= sg) gv = accG[st].f[r] * __expf(cumc[t] - cumc[sg]) * dtc[sg];
                split_store(&Ghi[t][st * 16 + l16], &Glo[t][st * 16 + l16], gv);
            }
        {   // stage XT[p][s] for this s-slab  (8 elems/thread)
            const int sl = tid >> 3;
            const int p0 = (tid & 7) * 8;
            const size_t row = tok0 + ss + sl;
            const float4 v0 = *(const float4*)(xBC + row * CONV_DIM + h * HEADDIM + p0);
            const float4 v1 = *(const float4*)(xBC + row * CONV_DIM + h * HEADDIM + p0 + 4);
            const float fv[8] = {v0.x, v0.y, v0.z, v0.w, v1.x, v1.y, v1.z, v1.w};
#pragma unroll
            for (int j = 0; j < 8; j++)
                split_store(&XThi[p0 + j][sl], &XTlo[p0 + j][sl], fv[j]);
        }
        __syncthreads();
        // Y += G_strip @ X_slab   (single K=32 step)
        BF16Frag gah, gal, bh2, bl2;
        buildA(gah, Ghi, wave * 16 + l16, hl);
        buildA(gal, Glo, wave * 16 + l16, hl);
#pragma unroll
        for (int nt = 0; nt < 4; nt++) {
            buildB(bh2, XThi, nt * 16 + l16, hl);
            buildB(bl2, XTlo, nt * 16 + l16, hl);
            wmma_bf16x3(accY[nt].v, gah, gal, bh2, bl2);
        }
        __syncthreads();
    }

    // ---------- epilogue: + D*x, store ----------
    const float Dh = Dp[h];
#pragma unroll
    for (int nt = 0; nt < 4; nt++) {
        const int p = nt * 16 + l16;
#pragma unroll
        for (int r = 0; r < 8; r++) {
            const int t = wave * 16 + hl * 8 + r;
            const float xv = xBC[(tok0 + t) * CONV_DIM + h * HEADDIM + p];
            y[(tok0 + t) * D_INNER + h * HEADDIM + p] = accY[nt].f[r] + Dh * xv;
        }
    }
}

// =====================================================================
// y = y * silu(z); rmsnorm over D_INNER; * norm_w   (in-place on y)
// =====================================================================
__global__ void __launch_bounds__(256)
gate_rmsnorm_kernel(const float* __restrict__ zx, const float* __restrict__ norm_w,
                    float* __restrict__ y)
{
    const int tok = blockIdx.x;
    const int tid = threadIdx.x;
    __shared__ float red[256];
    float vals[6];
    float ss = 0.f;
#pragma unroll
    for (int j = 0; j < 6; j++) {
        const int idx = j * 256 + tid;
        const float z = zx[(size_t)tok * D_IN_PROJ + idx];
        const float g = z / (1.f + __expf(-z));
        const float v = y[(size_t)tok * D_INNER + idx] * g;
        vals[j] = v;
        ss += v * v;
    }
    red[tid] = ss;
    __syncthreads();
    for (int s = 128; s > 0; s >>= 1) {
        if (tid < s) red[tid] += red[tid + s];
        __syncthreads();
    }
    const float rinv = rsqrtf(red[0] / (float)D_INNER + EPS);
#pragma unroll
    for (int j = 0; j < 6; j++) {
        const int idx = j * 256 + tid;
        y[(size_t)tok * D_INNER + idx] = vals[j] * rinv * norm_w[idx];
    }
}

// =====================================================================
extern "C" void kernel_launch(void* const* d_in, const int* in_sizes, int n_in,
                              void* d_out, int out_size, void* d_ws, size_t ws_size,
                              hipStream_t stream) {
    (void)in_sizes; (void)n_in; (void)out_size; (void)ws_size;
    const float* feature    = (const float*)d_in[0];
    const float* in_proj_w  = (const float*)d_in[1];
    const float* conv_w     = (const float*)d_in[2];
    const float* conv_b     = (const float*)d_in[3];
    const float* dt_bias    = (const float*)d_in[4];
    const float* A_log      = (const float*)d_in[5];
    const float* Dv         = (const float*)d_in[6];
    const float* norm_w     = (const float*)d_in[7];
    const float* out_proj_w = (const float*)d_in[8];
    const float* gate1      = (const float*)d_in[9];
    float* out = (float*)d_out;

    // workspace layout (floats)
    float* ws   = (float*)d_ws;
    float* zx   = ws;                                                  // [NTOK, 3352]
    float* xBC  = zx   + (size_t)NTOK * D_IN_PROJ;                     // [NTOK, 1792]
    float* dtv  = xBC  + (size_t)NTOK * CONV_DIM;                      // [NTOK, 24]
    float* ldA  = dtv  + (size_t)NTOK * NHEADS;                        // [NTOK, 24]
    float* cum  = ldA  + (size_t)NTOK * NHEADS;                        // [96*16, 128]
    float* Sbuf = cum  + (size_t)BATCH * NHEADS * NCHUNK * Q;          // [96*16, 64,128]
    float* y    = Sbuf + (size_t)BATCH * NHEADS * NCHUNK * HEADDIM * D_STATE; // [NTOK,1536]

    // 1) in_proj GEMM (WMMA)
    dim3 g1((D_IN_PROJ + 63) / 64, NTOK / 32);
    gemm_bf16x3_wmma<<<g1, 256, 0, stream>>>(feature, in_proj_w, zx,
                                             D_IN_PROJ, D_MODEL, nullptr);
    // 2) dt / log(dA)
    dt_da_kernel<<<(NTOK * NHEADS + 255) / 256, 256, 0, stream>>>(zx, dt_bias, A_log,
                                                                  dtv, ldA);
    // 3) depthwise conv + silu
    conv_silu_kernel<<<(NTOK * CONV_DIM + 255) / 256, 256, 0, stream>>>(zx, conv_w,
                                                                        conv_b, xBC);
    // 4) chunked scan: cumsum -> chunk states (WMMA) -> 16-step recurrence
    //    -> chunk outputs (WMMA)
    chunk_cumsum_kernel<<<(BATCH * NHEADS * NCHUNK + 255) / 256, 256, 0, stream>>>(ldA, cum);
    chunk_state_kernel<<<BATCH * NHEADS * NCHUNK, 256, 0, stream>>>(xBC, dtv, cum, Sbuf);
    state_recurrence_kernel<<<BATCH * NHEADS, 256, 0, stream>>>(cum, Sbuf);
    chunk_output_kernel<<<BATCH * NHEADS * NCHUNK, 256, 0, stream>>>(xBC, dtv, cum, Sbuf,
                                                                     Dv, y);
    // 5) gated RMSNorm (in-place on y)
    gate_rmsnorm_kernel<<<NTOK, 256, 0, stream>>>(zx, norm_w, y);
    // 6) out_proj GEMM (WMMA) with fused sigmoid(gate1)
    dim3 g2(D_MODEL / 64, NTOK / 32);
    gemm_bf16x3_wmma<<<g2, 256, 0, stream>>>(y, out_proj_w, out,
                                             D_MODEL, D_INNER, gate1);
}
